// OptimizedLightningIndexer_57604101374284
// MI455X (gfx1250) — compile-verified, matrix operand order checked by reading, not run on hardware
//
#include <hip/hip_runtime.h>
#include <hip/hip_bf16.h>
#include <stddef.h>

// ---------------------------------------------------------------------------
// Problem constants (from reference)
// ---------------------------------------------------------------------------
#define BATCH   2
#define SEQ     4096
#define DMODEL  1024
#define HEADS   2
#define IDXD    32
#define NCOLS   (HEADS * IDXD + IDXD + HEADS)   // 64 + 32 + 2 = 98

typedef _Float16 v16h __attribute__((ext_vector_type(16)));
typedef _Float16 v8h  __attribute__((ext_vector_type(8)));
typedef float    v8f  __attribute__((ext_vector_type(8)));

static __device__ inline v16h pack16(v8h lo, v8h hi) {
    v16h r;
#pragma unroll
    for (int i = 0; i < 8; ++i) { r[i] = lo[i]; r[i + 8] = hi[i]; }
    return r;
}

// ---------------------------------------------------------------------------
// Phase 1: projections  Q = x@Wq, K = x@Wk, w = x@Ww
// One block per (b,t) row. x row staged in LDS once; 98 output columns,
// one per thread, accumulated in f32 over the K=1024 contraction (kept in
// f32 for accuracy; total is only ~1.6 GFLOP). Q/K emitted as f16 in the
// WMMA-friendly layouts, w kept f32.
//   Qws: [b][h][t][32] f16,  Kws: [b][t][32] f16,  Wws: [b][t][2] f32
// ---------------------------------------------------------------------------
__global__ __launch_bounds__(128)
void lightning_proj(const float* __restrict__ x,
                    const float* __restrict__ Wq,
                    const float* __restrict__ Wk,
                    const float* __restrict__ Ww,
                    _Float16* __restrict__ Qws,
                    _Float16* __restrict__ Kws,
                    float* __restrict__ Wws)
{
    __shared__ float xs[DMODEL];

    const int row = blockIdx.x;               // b*SEQ + t
    const int c   = threadIdx.x;              // 0..127 (0..97 active for math)
    const float* xr = x + (size_t)row * DMODEL;

    // cooperative 16B-wide stage of the x row into LDS
    const float4* xr4 = (const float4*)xr;
#pragma unroll
    for (int k4 = c; k4 < DMODEL / 4; k4 += 128)
        ((float4*)xs)[k4] = xr4[k4];
    __syncthreads();

    if (c < NCOLS) {
        const float* wp;
        int stride;
        if (c < HEADS * IDXD)            { wp = Wq + c;                 stride = HEADS * IDXD; }
        else if (c < HEADS * IDXD + IDXD){ wp = Wk + (c - HEADS*IDXD);  stride = IDXD; }
        else                             { wp = Ww + (c - HEADS*IDXD - IDXD); stride = HEADS; }

        float acc = 0.0f;
#pragma unroll 8
        for (int k = 0; k < DMODEL; ++k)
            acc = fmaf(xs[k], wp[(size_t)k * stride], acc);

        const int b = row >> 12;              // SEQ = 4096
        const int t = row & (SEQ - 1);
        if (c < HEADS * IDXD) {
            const int h = c >> 5, d = c & 31;
            Qws[(((size_t)(b * HEADS + h)) * SEQ + t) * IDXD + d] = (_Float16)acc;
        } else if (c < HEADS * IDXD + IDXD) {
            Kws[((size_t)row) * IDXD + (c - HEADS * IDXD)] = (_Float16)acc;
        } else {
            Wws[(size_t)row * HEADS + (c - HEADS * IDXD - IDXD)] = acc;
        }
    }
}

// ---------------------------------------------------------------------------
// Phase 2: fused  out[b,t,s] = sum_h w[b,t,h] * relu( q[b,t,h,:] . k[b,s,:] )
//
// One wave32 per (t-tile x 4 s-tiles). Per head the 16x16 (t,s) dot tile is
// exactly one v_wmma_f32_16x16x32_f16 (K=32). ReLU + per-row head weight +
// head-sum happen in the f32 accumulator registers; only the final f32 tile
// is written (output write of 128 MB is the bandwidth floor).
//
// A-fragment (16x32 f16): lanes 0-15 row M=lane hold K 0-7 & 16-23;
// lanes 16-31 row M=lane-16 hold K 8-15 & 24-31. B mirrors with N=lane.
// C/D: VGPR r -> row r (lanes 0-15) or r+8 (lanes 16-31), col = lane&15.
// ---------------------------------------------------------------------------
#define STILES_PER_WAVE 4
#define WAVES_PER_BLOCK 8

__global__ __launch_bounds__(256)
void lightning_score(const _Float16* __restrict__ Q,
                     const _Float16* __restrict__ K,
                     const float* __restrict__ W,
                     float* __restrict__ out)
{
    const int wave = blockIdx.x * WAVES_PER_BLOCK + (threadIdx.x >> 5);
    const int lane = threadIdx.x & 31;

    // wave -> (b, t-tile, s-group of 4 tiles); 2 * 256 * 64 waves total
    const int sg = wave & 63;
    const int tt = (wave >> 6) & 255;
    const int b  = wave >> 14;

    const int t0   = tt * 16;
    const int half = lane >> 4;      // 0: lanes 0-15, 1: lanes 16-31
    const int lrow = lane & 15;
    const int kA0  = half * 8;       // K-chunk base per half-wave

    // ---- A fragments for both heads (reused across 4 s-tiles) ----
    v16h afrag[HEADS];
#pragma unroll
    for (int h = 0; h < HEADS; ++h) {
        const _Float16* qp =
            Q + (((size_t)(b * HEADS + h)) * SEQ + (t0 + lrow)) * IDXD + kA0;
        v8h lo = *(const v8h*)(qp);
        v8h hi = *(const v8h*)(qp + 16);
        afrag[h] = pack16(lo, hi);
    }

    // ---- per-row head weights matching the C/D register layout ----
    float wreg[HEADS][8];
#pragma unroll
    for (int h = 0; h < HEADS; ++h)
#pragma unroll
        for (int r = 0; r < 8; ++r)
            wreg[h][r] =
                W[((size_t)b * SEQ + t0 + r + half * 8) * HEADS + h];

    const _Float16* kbase = K + (size_t)b * SEQ * IDXD;

#pragma unroll
    for (int j = 0; j < STILES_PER_WAVE; ++j) {
        const int s0 = (sg * STILES_PER_WAVE + j) * 16;

        const _Float16* kp = kbase + (size_t)(s0 + lrow) * IDXD + kA0;
        if (j + 1 < STILES_PER_WAVE)   // hint next B tile (global_prefetch_b8)
            __builtin_prefetch(kp + 16 * IDXD, 0, 0);

        v8h blo = *(const v8h*)(kp);
        v8h bhi = *(const v8h*)(kp + 16);
        v16h bfrag = pack16(blo, bhi);

        v8f acc = {};
#pragma unroll
        for (int h = 0; h < HEADS; ++h) {
            v8f czero = {};
            v8f d = __builtin_amdgcn_wmma_f32_16x16x32_f16(
                        /*neg_a=*/false, afrag[h],
                        /*neg_b=*/false, bfrag,
                        /*c_mod=*/(short)0, czero,
                        /*reuse_a=*/false, /*reuse_b=*/false);
#pragma unroll
            for (int r = 0; r < 8; ++r)
                acc[r] += fmaxf(d[r], 0.0f) * wreg[h][r];
        }

        float* op = out + (((size_t)b * SEQ + t0 + half * 8) * SEQ) + s0 + lrow;
#pragma unroll
        for (int r = 0; r < 8; ++r)
            op[(size_t)r * SEQ] = acc[r];
    }
}

// ---------------------------------------------------------------------------
// Host-side launcher
// ---------------------------------------------------------------------------
extern "C" void kernel_launch(void* const* d_in, const int* in_sizes, int n_in,
                              void* d_out, int out_size, void* d_ws, size_t ws_size,
                              hipStream_t stream) {
    (void)in_sizes; (void)n_in; (void)out_size; (void)ws_size;

    const float* x  = (const float*)d_in[0];
    const float* Wq = (const float*)d_in[1];
    const float* Wk = (const float*)d_in[2];
    const float* Ww = (const float*)d_in[3];
    float* out = (float*)d_out;

    // workspace carving: Q f16 (1 MB) | K f16 (512 KB) | w f32 (64 KB)
    char* ws = (char*)d_ws;
    _Float16* Qws = (_Float16*)(ws);
    _Float16* Kws = (_Float16*)(ws + (size_t)BATCH * HEADS * SEQ * IDXD * sizeof(_Float16));
    float*    Wws = (float*)   (ws + (size_t)BATCH * HEADS * SEQ * IDXD * sizeof(_Float16)
                                   + (size_t)BATCH * SEQ * IDXD * sizeof(_Float16));

    // Phase 1: one block per (b,t) row
    lightning_proj<<<BATCH * SEQ, 128, 0, stream>>>(x, Wq, Wk, Ww, Qws, Kws, Wws);

    // Phase 2: 2 * 256 t-tiles * 64 s-groups waves / 8 waves per block
    const int total_waves = BATCH * (SEQ / 16) * ((SEQ / 16) / STILES_PER_WAVE);
    lightning_score<<<total_waves / WAVES_PER_BLOCK, 32 * WAVES_PER_BLOCK, 0, stream>>>(
        Qws, Kws, Wws, out);
}
